// R_TuckER_85830626443562
// MI455X (gfx1250) — compile-verified, matrix operand order checked by reading, not run on hardware
//
#include <hip/hip_runtime.h>
#include <hip/hip_bf16.h>

typedef __attribute__((ext_vector_type(16))) __bf16        v16bf;
typedef __attribute__((ext_vector_type(8)))  float         v8f;
typedef __attribute__((ext_vector_type(8)))  unsigned int  v8u;
typedef __attribute__((ext_vector_type(4)))  unsigned int  v4u;
typedef __attribute__((ext_vector_type(8)))  int           v8i;
typedef __attribute__((ext_vector_type(4)))  int           v4i;

#define RANK   200
#define KPAD   224              /* 7 * 32 */
#define KPADW  112              /* KPAD/2 uints (bf16 pairs) */
#define BB     512
#define EE     100000
#define ETILES (EE / 16)        /* 6250 */
#define IJ     (RANK * RANK)    /* 40000 */
#define IJW    (IJ / 2)         /* 20000 uints per transposed-core row */
#define ZSPLIT 5                /* 8000 ij = 40 core rows per split (row-aligned) */
#define NT_PER_BLK 8

/* stage-1 B chunk: 10 WMMA steps = 320 ij = 160 uints per core column */
#define B1_W    160
#define B1_ROW  165             /* 160 + 160/32 TDM pad dwords -> odd LDS stride */
#define F2_ROW  119             /* 112 + 112/16 TDM pad dwords -> odd LDS stride */

/* TDM pad configs for D# group1 word0: pad_enable | pad_interval | pad_amount=1 */
#define PAD_32x1 ((1u << 20) | (4u << 22))   /* every 32 dwords, +1 dword */
#define PAD_16x1 ((1u << 20) | (3u << 22))   /* every 16 dwords, +1 dword */

// ---- bf16 helpers (manual RNE, used only in cold staging/prep paths) ------
__device__ __forceinline__ unsigned int f2bf(float f) {
    unsigned int u = __builtin_bit_cast(unsigned int, f);
    return (u + 0x7fffu + ((u >> 16) & 1u)) >> 16;
}
__device__ __forceinline__ unsigned int pack2bf(float lo, float hi) {
    return f2bf(lo) | (f2bf(hi) << 16);
}
// ISA 16-bit A(16x32)/B(32x16) layout: VGPR pair p, lane-half h -> K value
__device__ __forceinline__ int kmap(int p, int h) {
    return (p < 4) ? (8 * h + 2 * p) : (16 + 8 * h + 2 * (p - 4));
}
__device__ __forceinline__ v8f bfwmma(v8u a, v8u b, v8f c) {
    return __builtin_amdgcn_wmma_f32_16x16x32_bf16(
        false, __builtin_bit_cast(v16bf, a),
        false, __builtin_bit_cast(v16bf, b),
        (short)0, c, false, false);
}

// ---- Tensor Data Mover: 2D tile (4-byte elements) global -> LDS -----------
// D# per CDNA5 ISA 8.3/8.4: group0 = {count=1, lds_addr, global_addr, type=2},
// group1 = {data_size=4B + pad cfg, tensor dims, tile dims, dim0 stride}.
// This toolchain exposes the 6-arg builtin: (g0, g1, g2, g3, g_extra, cpol).
__device__ __forceinline__ void tdm_load_2d(unsigned int lds_byte_off,
                                            const void* gptr,
                                            unsigned int tensor_d0,
                                            unsigned int tensor_d1,
                                            unsigned int tile_d0,
                                            unsigned int tile_d1,
                                            unsigned int stride_d0,
                                            unsigned int pad_cfg)
{
    const unsigned long long ga = (unsigned long long)(uintptr_t)gptr;
    v4u g0;
    g0[0] = 1u;                                               // count=1, user D#
    g0[1] = lds_byte_off;                                     // bits 63:32
    g0[2] = (unsigned int)ga;                                 // addr[31:0]
    g0[3] = (unsigned int)((ga >> 32) & 0x01ffffffu) | (2u << 30); // addr[56:32]|type=2
    v8i g1;
    g1[0] = (int)((2u << 16) | pad_cfg);                      // data_size=4B
    g1[1] = (int)((tensor_d0 & 0xffffu) << 16);               // tensor_dim0[15:0]
    g1[2] = (int)((tensor_d0 >> 16) | ((tensor_d1 & 0xffffu) << 16));
    g1[3] = (int)((tensor_d1 >> 16) | (tile_d0 << 16));       // tile_dim0
    g1[4] = (int)(tile_d1 & 0xffffu);                         // tile_dim1 (dim2=0)
    g1[5] = (int)stride_d0;                                   // tensor_dim0_stride
    g1[6] = 0;
    g1[7] = 0;
    const v4i gz4 = {0, 0, 0, 0};
    const v8i gz8 = {0, 0, 0, 0, 0, 0, 0, 0};
    __builtin_amdgcn_tensor_load_to_lds(g0, g1, gz4, gz4, gz8, 0);
}

// ---------------------------------------------------------------------------
// Prep A: F2 [E][200] f32 -> f2_bf [E][224] bf16 pairs, zero-padded K.
// ---------------------------------------------------------------------------
__global__ void tucker_prep_f2(const float* __restrict__ F2,
                               unsigned int* __restrict__ f2bf32)
{
    const long long idx = (long long)blockIdx.x * blockDim.x + threadIdx.x;
    if (idx >= (long long)EE * KPADW) return;
    const int e  = (int)(idx / KPADW);
    const int c2 = (int)(idx % KPADW) * 2;
    float a = 0.0f, b = 0.0f;
    if (c2 < RANK) {
        a = F2[(size_t)e * RANK + c2];
        b = F2[(size_t)e * RANK + c2 + 1];
    }
    f2bf32[idx] = pack2bf(a, b);
}

// ---------------------------------------------------------------------------
// Prep B: core [40000][200] f32 -> coreT [200][40000] bf16 pairs (transposed
// so consecutive-ij bf16 pairs are contiguous). LDS-tiled 32x32 transpose.
// ---------------------------------------------------------------------------
__global__ __launch_bounds__(256)
void tucker_prep_core(const float* __restrict__ core,
                      unsigned int* __restrict__ coreT32)
{
    __shared__ float tile[32][33];
    const int ij0 = blockIdx.x * 32;
    const int k0  = blockIdx.y * 32;
    const int tx  = threadIdx.x & 31;
    const int ty  = threadIdx.x >> 5;

#pragma unroll
    for (int q = 0; q < 4; ++q) {
        const int r = ij0 + ty + 8 * q;
        const int k = k0 + tx;
        tile[ty + 8 * q][tx] = (k < RANK) ? core[(size_t)r * RANK + k] : 0.0f;
    }
    __syncthreads();
#pragma unroll
    for (int q = 0; q < 4; ++q) {
        const int k = k0 + ty + 8 * q;
        if (k < RANK && tx < 16) {
            coreT32[(size_t)k * IJW + (ij0 >> 1) + tx] =
                pack2bf(tile[2 * tx][ty + 8 * q], tile[2 * tx + 1][ty + 8 * q]);
        }
    }
}

// ---------------------------------------------------------------------------
// Stage 1: w[b,k] = sum_{ij} (u[b,i]*v[b,j]) * core[ij,k]   (bf16 WMMA)
// A fragments built in-register with V_PK_MUL_BF16 from LDS-resident bf16
// u (broadcast pairs) / v (pairs). B slabs (16 core cols x 320 ij) arrive via
// double-buffered TENSOR_LOAD_TO_LDS; inner loop is pure LDS -> WMMA.
// ---------------------------------------------------------------------------
__global__ __launch_bounds__(32)
void tucker_bilinear_bf16(const int* __restrict__ subj, const int* __restrict__ rel,
                          const float* __restrict__ F0, const float* __restrict__ F1,
                          const unsigned int* __restrict__ coreT32,
                          float* __restrict__ w_part)
{
    __shared__ unsigned int lds_u32[16 * RANK + 2];    // bf16 broadcast pairs
    __shared__ unsigned int lds_v32[16 * (RANK / 2)];  // bf16 pairs
    __shared__ unsigned int lds_B[2 * 16 * B1_ROW];    // TDM double buffer

    const int kt   = blockIdx.x;   // 0..12
    const int bt   = blockIdx.y;   // 0..31
    const int z    = blockIdx.z;   // 0..4
    const int lane = threadIdx.x;

    for (int t = lane; t < 16 * RANK; t += 32) {
        const int mm = t / RANK, c = t % RANK;
        const unsigned int b = f2bf(F0[(size_t)subj[bt * 16 + mm] * RANK + c]);
        lds_u32[t] = b | (b << 16);
    }
    for (int t = lane; t < 16 * (RANK / 2); t += 32) {
        const int mm = t / (RANK / 2), c2 = (t % (RANK / 2)) * 2;
        const size_t base = (size_t)rel[bt * 16 + mm] * RANK + c2;
        lds_v32[t] = pack2bf(F1[base], F1[base + 1]);
    }
    __syncthreads();

    const int m   = lane & 15;
    const int h   = lane >> 4;
    const int n   = lane & 15;
    const int col = kt * 16 + n;

    const unsigned int ldsB0 = (unsigned int)(uintptr_t)(&lds_B[0]);
    const unsigned int* gB = coreT32 + (size_t)kt * 16 * IJW
                                     + (size_t)z * (IJ / ZSPLIT / 2);

    tdm_load_2d(ldsB0, gB, B1_W, 16, B1_W, 16, IJW, PAD_32x1);   // chunk 0

    int ib = z * (IJ / ZSPLIT / RANK);   // current core row (z*40)
    int j0 = 0;                          // column of K=0 within row (even)
    v8f acc = {};

    for (int c = 0; c < 25; ++c) {                       // 25 chunks x 10 WMMA
        if (c + 1 < 25) {
            tdm_load_2d(ldsB0 + ((c + 1) & 1) * (16 * B1_ROW * 4),
                        gB + (size_t)(c + 1) * B1_W,
                        B1_W, 16, B1_W, 16, IJW, PAD_32x1);
            __builtin_amdgcn_s_wait_tensorcnt(1);
        } else {
            __builtin_amdgcn_s_wait_tensorcnt(0);
        }
        const unsigned int* Bbuf = &lds_B[(c & 1) * (16 * B1_ROW)];

        for (int ls = 0; ls < 10; ++ls) {
            const unsigned int u0p = lds_u32[m * RANK + ib];
            const unsigned int u1p = lds_u32[m * RANK + ib + 1];  // straddle row
            v8u au, bu;
#pragma unroll
            for (int p = 0; p < 8; ++p) {
                const int K0 = kmap(p, h);
                const int j  = j0 + K0;
                const bool wr = (j >= RANK);        // pair never splits (RANK even)
                const int jj  = wr ? j - RANK : j;
                const unsigned int vv = lds_v32[m * (RANK / 2) + (jj >> 1)];
                const unsigned int up = wr ? u1p : u0p;
                unsigned int prod;
                asm("v_pk_mul_bf16 %0, %1, %2" : "=v"(prod) : "v"(up), "v"(vv));
                au[p] = prod;
                bu[p] = Bbuf[n * B1_ROW + ls * 16 + (ls >> 1) + (K0 >> 1)];
            }
            acc = bfwmma(au, bu, acc);
            j0 += 32;
            if (j0 >= RANK) { j0 -= RANK; ++ib; }
        }
    }

    float* wp = w_part + (size_t)z * BB * RANK;
    if (col < RANK) {
#pragma unroll
        for (int r = 0; r < 8; ++r)
            wp[(size_t)(bt * 16 + r + 8 * h) * RANK + col] = acc[r];
    }
}

// ---------------------------------------------------------------------------
// Stage 2: deterministic reduce of 5 K-split partials -> w_bf [512][224] bf16.
// ---------------------------------------------------------------------------
__global__ void tucker_reduce_w(const float* __restrict__ w_part,
                                unsigned int* __restrict__ wbf32)
{
    const int idx = blockIdx.x * blockDim.x + threadIdx.x;
    if (idx >= BB * KPADW) return;
    const int b = idx / KPADW, c2 = (idx % KPADW) * 2;
    float s0 = 0.0f, s1 = 0.0f;
    if (c2 < RANK) {
#pragma unroll
        for (int zz = 0; zz < ZSPLIT; ++zz) {
            s0 += w_part[(size_t)zz * BB * RANK + (size_t)b * RANK + c2];
            s1 += w_part[(size_t)zz * BB * RANK + (size_t)b * RANK + c2 + 1];
        }
    }
    wbf32[idx] = pack2bf(s0, s1);
}

// ---------------------------------------------------------------------------
// Stage 3: sigmoid(w @ F2^T) -> out [512][100000] f32 (memory bound: ~205MB
// stores ~ 8.8us @ 23.3 TB/s). 7 A-fragments pinned in VGPRs per block; F2
// tiles via double-buffered TENSOR_LOAD_TO_LDS with bank-conflict-free pad.
// ---------------------------------------------------------------------------
__global__ __launch_bounds__(32)
void tucker_logits_bf16(const unsigned int* __restrict__ wbf32,
                        const unsigned int* __restrict__ f2bf32,
                        float* __restrict__ out)
{
    __shared__ unsigned int lds_f2[2 * 16 * F2_ROW];

    const int bt   = blockIdx.y;
    const int lane = threadIdx.x;
    const int m    = lane & 15;
    const int h    = lane >> 4;
    const int n    = lane & 15;

    // Load the 7 A fragments (w row bt*16+m, all K chunks) once; reuse 8x.
    v8u afrag[7];
    const unsigned int* wrow = wbf32 + (size_t)(bt * 16 + m) * KPADW;
#pragma unroll
    for (int t = 0; t < 7; ++t)
#pragma unroll
        for (int p = 0; p < 8; ++p)
            afrag[t][p] = wrow[(t * 32 + kmap(p, h)) >> 1];

    const unsigned int lds0 = (unsigned int)(uintptr_t)(&lds_f2[0]);
    const int et0 = blockIdx.x * NT_PER_BLK;
    tdm_load_2d(lds0, f2bf32 + (size_t)et0 * 16 * KPADW,
                KPADW, 16, KPADW, 16, KPADW, PAD_16x1);

    for (int nt = 0; nt < NT_PER_BLK; ++nt) {
        const int et = et0 + nt;
        if (et >= ETILES) break;
        const bool more = (nt + 1 < NT_PER_BLK) && (et + 1 < ETILES);
        if (more) {
            tdm_load_2d(lds0 + ((nt + 1) & 1) * (16 * F2_ROW * 4),
                        f2bf32 + (size_t)(et + 1) * 16 * KPADW,
                        KPADW, 16, KPADW, 16, KPADW, PAD_16x1);
            __builtin_amdgcn_s_wait_tensorcnt(1);
        } else {
            __builtin_amdgcn_s_wait_tensorcnt(0);
        }
        const unsigned int* Bt = &lds_f2[(nt & 1) * (16 * F2_ROW)];

        v8f acc = {};
#pragma unroll
        for (int t = 0; t < 7; ++t) {
            v8u bu;
#pragma unroll
            for (int p = 0; p < 8; ++p)
                bu[p] = Bt[n * F2_ROW + t * 17 + (kmap(p, h) >> 1)];
            acc = bfwmma(afrag[t], bu, acc);
        }
#pragma unroll
        for (int r = 0; r < 8; ++r) {
            const float sg = 1.0f / (1.0f + __expf(-acc[r]));
            out[(size_t)(bt * 16 + r + 8 * h) * EE + (size_t)et * 16 + n] = sg;
        }
    }
}

// ---------------------------------------------------------------------------
// Inputs: subject_idx[i32,512], relation_idx[i32,512], core[f32,200^3],
// F0[f32,1e5x200], F1[f32,500x200], F2[f32,1e5x200].  Output f32 [512,1e5].
// Workspace: f2_bf 44.8M | coreT 16M | w_part 2.05M | w_bf 0.23M  (~63MB)
// ---------------------------------------------------------------------------
extern "C" void kernel_launch(void* const* d_in, const int* in_sizes, int n_in,
                              void* d_out, int out_size, void* d_ws, size_t ws_size,
                              hipStream_t stream)
{
    const int*   subj = (const int*)  d_in[0];
    const int*   rel  = (const int*)  d_in[1];
    const float* core = (const float*)d_in[2];
    const float* F0   = (const float*)d_in[3];
    const float* F1   = (const float*)d_in[4];
    const float* F2   = (const float*)d_in[5];
    float*       out  = (float*)d_out;

    char* ws = (char*)d_ws;
    unsigned int* f2bf32  = (unsigned int*)ws;
    unsigned int* coreT32 = (unsigned int*)(ws + (size_t)EE * KPADW * 4);
    float*        w_part  = (float*)(ws + (size_t)EE * KPADW * 4
                                        + (size_t)RANK * IJW * 4);
    unsigned int* wbf32   = (unsigned int*)((char*)w_part
                                        + (size_t)ZSPLIT * BB * RANK * 4);

    {
        const long long nthr = (long long)EE * KPADW;
        tucker_prep_f2<<<(unsigned)((nthr + 255) / 256), 256, 0, stream>>>(F2, f2bf32);
    }
    {
        dim3 g(IJ / 32, (RANK + 31) / 32);   // 1250 x 7
        tucker_prep_core<<<g, 256, 0, stream>>>(core, coreT32);
    }
    {
        dim3 g((RANK + 15) / 16, BB / 16, ZSPLIT);   // 13 x 32 x 5
        tucker_bilinear_bf16<<<g, 32, 0, stream>>>(subj, rel, F0, F1, coreT32, w_part);
    }
    tucker_reduce_w<<<(BB * KPADW + 255) / 256, 256, 0, stream>>>(w_part, wbf32);
    {
        dim3 g((ETILES + NT_PER_BLK - 1) / NT_PER_BLK, BB / 16);   // 782 x 32
        tucker_logits_bf16<<<g, 32, 0, stream>>>(wbf32, f2bf32, out);
    }
}
